// WCHeadGAPModelL_80668075753619
// MI455X (gfx1250) — matrix operand order repaired and checked by reading, not compiled, not run on hardware
//
#include <hip/hip_runtime.h>
#include <math.h>

typedef __attribute__((ext_vector_type(16))) __bf16 v16bf;
typedef __attribute__((ext_vector_type(8)))  __bf16 v8bf;
typedef __attribute__((ext_vector_type(8)))  float  v8f;

#define GNUM 40000
#define NPART 5
#define NT_TILES 2505            // ceil(G/16) + P padding tiles
#define PIDX_LEN (NT_TILES * 16)

// swizzled MLP weight segment offsets (halves), per part
#define O_L1IN  0
#define O_L1H0  12288
#define O_L1H1  28672
#define O_L1OUT 45056
#define O_L2IN  47104
#define O_L2H0  63488
#define O_L2H1  79872
#define O_L2OUT 96256
#define MLPW_PART 100352
#define CONVIN_SW 18432
#define CONVHID_SW 36864

__device__ inline unsigned short f2bf(float f) {
  unsigned int u = __float_as_uint(f);
  unsigned int r = u + 0x7fffu + ((u >> 16) & 1u);
  return (unsigned short)(r >> 16);
}
__device__ inline float bf2f(unsigned short h) {
  return __uint_as_float(((unsigned int)h) << 16);
}
__device__ inline v8f zero8() {
  v8f z = {0.f, 0.f, 0.f, 0.f, 0.f, 0.f, 0.f, 0.f};
  return z;
}
__device__ inline v8bf zbf8() {
  v8bf z;
#pragma unroll
  for (int i = 0; i < 8; ++i) z[i] = (__bf16)0.0f;
  return z;
}
__device__ inline v8f wmma_bf16(v16bf a, v16bf b, v8f c) {
  return __builtin_amdgcn_wmma_f32_16x16x32_bf16(false, a, false, b, (short)0, c,
                                                 false, false);
}
__device__ inline v16bf cat16(v8bf lo, v8bf hi) {
  return __builtin_shufflevector(lo, hi, 0, 1, 2, 3, 4, 5, 6, 7, 8, 9, 10, 11, 12,
                                 13, 14, 15);
}
// A fragment (16x32 bf16, contiguous k = 16*hi + s) from an LDS row-major [16][128] buffer
__device__ inline v16bf load_a_lds(const unsigned short* lds, int kbase, int lane) {
  int m = lane & 15, hi = lane >> 4;
  const unsigned short* row = lds + m * 128;
  v8bf lo = *(const v8bf*)(row + kbase + hi * 16);
  v8bf hj = *(const v8bf*)(row + kbase + hi * 16 + 8);
  return cat16(lo, hj);
}

// ---------------------------------------------------------------- init
__global__ __launch_bounds__(256) void init_kernel(int* counts, int* cursors,
                                                   int* tile_part, int* pidx,
                                                   unsigned short* uvmap) {
  int i = blockIdx.x * 256 + threadIdx.x;  // grid covers 2097152
  uvmap[i] = 0;
  if (i < NPART) { counts[i] = 0; cursors[i] = 0; }
  if (i < NT_TILES) tile_part[i] = -1;
  if (i < PIDX_LEN) pidx[i] = -1;
}

// ---------------------------------------------------------------- bucketize
__global__ __launch_bounds__(256) void count_kernel(const int* gs_part, int* counts) {
  int g = blockIdx.x * 256 + threadIdx.x;
  if (g >= GNUM) return;
  atomicAdd(&counts[gs_part[g]], 1);
}

__global__ void offsets_kernel(const int* counts, int* offs, int* tile_part) {
  int acc = 0;
  for (int p = 0; p < NPART; ++p) {
    offs[p] = acc;
    int c = counts[p];
    int ct = (c + 15) >> 4;
    int t0 = acc >> 4;
    for (int t = 0; t < ct; ++t) tile_part[t0 + t] = p;
    acc += ct << 4;
  }
  offs[NPART] = acc;
}

__global__ __launch_bounds__(256) void scatter_kernel(const int* gs_part,
                                                      const int* offs, int* cursors,
                                                      int* pidx) {
  int g = blockIdx.x * 256 + threadIdx.x;
  if (g >= GNUM) return;
  int p = gs_part[g];
  int pos = atomicAdd(&cursors[p], 1);
  pidx[offs[p] + pos] = g;
}

// ---------------------------------------------------------------- MLP weight swizzle
__global__ __launch_bounds__(512) void mlp_prep_kernel(
    const float* w1_in, const float* w1_hid, const float* w1_out, const float* w2_in,
    const float* w2_hid, const float* w2_out, unsigned short* out) {
  int bid = blockIdx.x;  // 0..979
  int p = bid / 196, r = bid % 196;
  int s = threadIdx.x;
  int lane = s >> 4, j = s & 15;
  int hi = lane >> 4, nl = lane & 15;
  int segO, r2, nt, kc;
  float v = 0.f;
  if (r < 24) {  // L1 in: K=96 (3 chunks) x N=128 (8 tiles)
    segO = O_L1IN; r2 = r; nt = r2 / 3; kc = r2 % 3;
    int k = kc * 32 + hi * 16 + j, n = nt * 16 + nl;
    v = w1_in[((size_t)p * 96 + k) * 128 + n];
  } else if (r < 88) {  // L1 hidden x2: 128x128
    int l = (r - 24) / 32; r2 = (r - 24) % 32;
    segO = O_L1H0 + l * 16384; nt = r2 / 4; kc = r2 % 4;
    int k = kc * 32 + hi * 16 + j, n = nt * 16 + nl;
    v = w1_hid[(((size_t)p * 2 + l) * 128 + k) * 128 + n];
  } else if (r < 92) {  // L1 out: 128x11 (pad N to 16)
    r2 = r - 88; segO = O_L1OUT; nt = 0; kc = r2;
    int k = kc * 32 + hi * 16 + j, n = nl;
    v = (n < 11) ? w1_out[((size_t)p * 128 + k) * 11 + n] : 0.f;
  } else if (r < 124) {  // L2 in: 110x128 (pad K to 128)
    r2 = r - 92; segO = O_L2IN; nt = r2 / 4; kc = r2 % 4;
    int k = kc * 32 + hi * 16 + j, n = nt * 16 + nl;
    v = (k < 110) ? w2_in[((size_t)p * 110 + k) * 128 + n] : 0.f;
  } else if (r < 188) {  // L2 hidden x2
    int l = (r - 124) / 32; r2 = (r - 124) % 32;
    segO = O_L2H0 + l * 16384; nt = r2 / 4; kc = r2 % 4;
    int k = kc * 32 + hi * 16 + j, n = nt * 16 + nl;
    v = w2_hid[(((size_t)p * 2 + l) * 128 + k) * 128 + n];
  } else {  // L2 out: 128x32
    r2 = r - 188; segO = O_L2OUT; nt = r2 / 4; kc = r2 % 4;
    int k = kc * 32 + hi * 16 + j, n = nt * 16 + nl;
    v = w2_out[((size_t)p * 128 + k) * 32 + n];
  }
  out[(size_t)p * MLPW_PART + segO + r2 * 512 + s] = f2bf(v);
}

// ---------------------------------------------------------------- conv weight swizzle
__global__ __launch_bounds__(512) void conv_prep_kernel(const float* w_in,
                                                        const float* w_hid,
                                                        unsigned short* out) {
  int bid = blockIdx.x;  // 0..251
  int s = threadIdx.x;
  int lane = s >> 4, j = s & 15;
  int hi = lane >> 4, nl = lane & 15;
  float v;
  size_t oidx;
  if (bid < 36) {  // conv_in: Cin=32 (KC=1), tap-major: b2 = tap*4+nt
    int tap = bid / 4, nt = bid % 4;
    int o = nt * 16 + nl, i = hi * 16 + j;
    v = w_in[((size_t)o * 32 + i) * 9 + tap];
    oidx = (size_t)bid * 512 + s;
  } else {  // hidden: Cin=64 (KC=2), b2 = (tap*2+kc)*4+nt
    int r = bid - 36;
    int layer = r / 72, b2 = r % 72;
    int nt = b2 & 3, q = b2 >> 2;
    int kc = q & 1, tap = q >> 1;
    int o = nt * 16 + nl, i = kc * 32 + hi * 16 + j;
    v = w_hid[(((size_t)layer * 64 + o) * 64 + i) * 9 + tap];
    oidx = (size_t)CONVIN_SW + (size_t)layer * CONVHID_SW + (size_t)b2 * 512 + s;
  }
  out[oidx] = f2bf(v);
}

// ---------------------------------------------------------------- MLP layer (WMMA)
template <int KC, int NT, bool LEAKY>
__device__ inline void mlp_layer(const unsigned short* src, unsigned short* dst,
                                 const unsigned short* wsw, const float* bias,
                                 int lane) {
  int hi = lane >> 4, n = lane & 15;
#pragma unroll
  for (int nt = 0; nt < NT; ++nt) {
    v8f d = zero8();
#pragma unroll
    for (int kc = 0; kc < KC; ++kc) {
      v16bf a = load_a_lds(src, kc * 32, lane);
      v16bf b = *(const v16bf*)(wsw + ((size_t)(nt * KC + kc)) * 512 + lane * 16);
      d = wmma_bf16(a, b, d);
    }
    float bn = bias[nt * 16 + n];
#pragma unroll
    for (int r = 0; r < 8; ++r) {
      float v = d[r] + bn;
      if (LEAKY) v = v > 0.f ? v : 0.01f * v;
      dst[(r + 8 * hi) * 128 + nt * 16 + n] = f2bf(v);
    }
  }
}

// ---------------------------------------------------------------- MLP kernel
__global__ __launch_bounds__(32) void mlp_kernel(
    const int* tile_part, const int* pidx, const float* latent_z,
    const float* latent_f, const float* cano_xyz, const unsigned short* mlpw,
    const float* b1_in, const float* b1_hid, const float* b1_out, const float* b2_in,
    const float* b2_hid, const float* b2_out, const int* uv_idx,
    unsigned short* uvmap) {
  __shared__ __align__(16) unsigned short bufX[16 * 128];
  __shared__ __align__(16) unsigned short bufA[16 * 128];
  __shared__ __align__(16) unsigned short bufB[16 * 128];
  int t = blockIdx.x, lane = threadIdx.x;
  int part = tile_part[t];
  if (part < 0) return;
  const unsigned short* W = mlpw + (size_t)part * MLPW_PART;

  // build x1 = [f_part | z | xyz | attrs(later) | zeros]
  for (int i = lane; i < 2048; i += 32) {
    int m = i >> 7, c = i & 127;
    int g = pidx[t * 16 + m];
    float v = 0.f;
    if (g >= 0) {
      if (c < 64) v = latent_f[part * 64 + c];
      else if (c < 96) v = latent_z[(size_t)g * 32 + (c - 64)];
      else if (c < 99) v = cano_xyz[(size_t)g * 3 + (c - 96)];
    }
    bufX[i] = f2bf(v);
  }
  __syncthreads();

  // MLP1: 96 -> 128 -> 128 -> 128 -> 11
  mlp_layer<3, 8, true>(bufX, bufA, W + O_L1IN, b1_in + part * 128, lane);
  __syncthreads();
  mlp_layer<4, 8, true>(bufA, bufB, W + O_L1H0, b1_hid + (part * 2 + 0) * 128, lane);
  __syncthreads();
  mlp_layer<4, 8, true>(bufB, bufA, W + O_L1H1, b1_hid + (part * 2 + 1) * 128, lane);
  __syncthreads();
  {  // out layer -> attrs into bufX cols 99..109 (no activation)
    int hi = lane >> 4, n = lane & 15;
    v8f d = zero8();
#pragma unroll
    for (int kc = 0; kc < 4; ++kc) {
      v16bf a = load_a_lds(bufA, kc * 32, lane);
      v16bf b = *(const v16bf*)(W + O_L1OUT + (size_t)kc * 512 + lane * 16);
      d = wmma_bf16(a, b, d);
    }
    if (n < 11) {
      float bn = b1_out[part * 11 + n];
#pragma unroll
      for (int r = 0; r < 8; ++r)
        bufX[(r + 8 * hi) * 128 + 99 + n] = f2bf(d[r] + bn);
    }
  }
  __syncthreads();

  // MLP2: 110(pad128) -> 128 -> 128 -> 128 -> 32
  mlp_layer<4, 8, true>(bufX, bufA, W + O_L2IN, b2_in + part * 128, lane);
  __syncthreads();
  mlp_layer<4, 8, true>(bufA, bufB, W + O_L2H0, b2_hid + (part * 2 + 0) * 128, lane);
  __syncthreads();
  mlp_layer<4, 8, true>(bufB, bufA, W + O_L2H1, b2_hid + (part * 2 + 1) * 128, lane);
  __syncthreads();
  {  // out layer -> scatter h (bf16) into UV map
    int hi = lane >> 4, n = lane & 15;
#pragma unroll
    for (int nt = 0; nt < 2; ++nt) {
      v8f d = zero8();
#pragma unroll
      for (int kc = 0; kc < 4; ++kc) {
        v16bf a = load_a_lds(bufA, kc * 32, lane);
        v16bf b =
            *(const v16bf*)(W + O_L2OUT + (size_t)(nt * 4 + kc) * 512 + lane * 16);
        d = wmma_bf16(a, b, d);
      }
      float bn = b2_out[part * 32 + nt * 16 + n];
#pragma unroll
      for (int r = 0; r < 8; ++r) {
        int m = r + 8 * hi;
        int g = pidx[t * 16 + m];
        if (g >= 0)
          uvmap[(size_t)uv_idx[g] * 32 + nt * 16 + n] = f2bf(d[r] + bn);
      }
    }
  }
}

// ---------------------------------------------------------------- conv (WMMA implicit GEMM)
template <int KC>  // Cin = KC*32
__global__ __launch_bounds__(128) void conv_kernel(const unsigned short* act,
                                                   const unsigned short* wsw,
                                                   const float* bias, float* out) {
  int lane = threadIdx.x & 31;
  int tileId = blockIdx.x * 4 + (threadIdx.x >> 5);  // 16384 tiles
  int nt = tileId & 3, xt = (tileId >> 2) & 15, y = tileId >> 6;
  int x0 = xt * 16;
  int hi = lane >> 4, n = lane & 15;
  const int Cin = KC * 32;
  v8f d = zero8();
#pragma unroll
  for (int tap = 0; tap < 9; ++tap) {
    int dy = tap / 3 - 1, dx = tap % 3 - 1;
    int py = y + dy, px = x0 + n + dx;
    bool valid = (py >= 0) & (py < 256) & (px >= 0) & (px < 256);
    const unsigned short* rowp = act + ((size_t)(py * 256 + px)) * Cin;
#pragma unroll
    for (int kc = 0; kc < KC; ++kc) {
      v8bf lo, hj;
      if (valid) {
        lo = *(const v8bf*)(rowp + kc * 32 + hi * 16);
        hj = *(const v8bf*)(rowp + kc * 32 + hi * 16 + 8);
      } else {
        lo = zbf8();
        hj = zbf8();
      }
      v16bf a = cat16(lo, hj);
      v16bf b =
          *(const v16bf*)(wsw + ((size_t)((tap * KC + kc) * 4 + nt)) * 512 + lane * 16);
      d = wmma_bf16(a, b, d);
    }
  }
  float bn = bias[nt * 16 + n];
#pragma unroll
  for (int r = 0; r < 8; ++r) {
    int m = r + 8 * hi;
    out[((size_t)(y * 256 + x0 + m)) * 64 + nt * 16 + n] = d[r] + bn;
  }
}

// ---------------------------------------------------------------- instance-norm stats
__global__ __launch_bounds__(256) void stats_kernel(const float* y, float* stats) {
  int c = blockIdx.x, tid = threadIdx.x;
  double s = 0.0, s2 = 0.0;
  for (int i = tid; i < 65536; i += 256) {
    float v = y[(size_t)i * 64 + c];
    s += v;
    s2 += (double)v * (double)v;
  }
  __shared__ double sh[256], sh2[256];
  sh[tid] = s;
  sh2[tid] = s2;
  __syncthreads();
  for (int o = 128; o > 0; o >>= 1) {
    if (tid < o) { sh[tid] += sh[tid + o]; sh2[tid] += sh2[tid + o]; }
    __syncthreads();
  }
  if (tid == 0) {
    double m = sh[0] / 65536.0;
    double var = sh2[0] / 65536.0 - m * m;
    stats[c] = (float)m;
    stats[64 + c] = (float)(1.0 / sqrt(var + 1e-5));
  }
}

// ---------------------------------------------------------------- IN / relu / bf16 prep
__global__ __launch_bounds__(256) void prep_kernel(const float* y, const float* stats,
                                                   int mode, unsigned short* out) {
  size_t i = (size_t)blockIdx.x * 256 + threadIdx.x;  // 4194304 elems exact
  int c = (int)(i & 63);
  float v = y[i];
  if (mode & 1) v = (v - stats[c]) * stats[64 + c];
  if (mode & 2) v = fmaxf(v, 0.f);
  out[i] = f2bf(v);
}

// ---------------------------------------------------------------- conv_out + sigmoid
__global__ __launch_bounds__(256) void convout_kernel(const unsigned short* act,
                                                      const float* w, const float* b,
                                                      float* out) {
  int i = blockIdx.x * 256 + threadIdx.x;  // 3*65536 exact
  int c = i >> 16, p = i & 65535;
  int y = p >> 8, x = p & 255;
  float s = b[c];
  for (int tap = 0; tap < 9; ++tap) {
    int py = y + tap / 3 - 1, px = x + tap % 3 - 1;
    if (py < 0 || py >= 256 || px < 0 || px >= 256) continue;
    const unsigned short* row = act + ((size_t)(py * 256 + px)) * 64;
#pragma unroll 8
    for (int k = 0; k < 64; ++k)
      s += bf2f(row[k]) * w[((size_t)c * 64 + k) * 9 + tap];
  }
  out[i] = 1.f / (1.f + expf(-s));
}

// ---------------------------------------------------------------- host
extern "C" void kernel_launch(void* const* d_in, const int* in_sizes, int n_in,
                              void* d_out, int out_size, void* d_ws, size_t ws_size,
                              hipStream_t stream) {
  (void)in_sizes; (void)n_in; (void)out_size; (void)ws_size;
  const float* latent_z = (const float*)d_in[0];
  const float* latent_f = (const float*)d_in[1];
  const float* cano_xyz = (const float*)d_in[2];
  const float* w1_in = (const float*)d_in[3];
  const float* b1_in = (const float*)d_in[4];
  const float* w1_hid = (const float*)d_in[5];
  const float* b1_hid = (const float*)d_in[6];
  const float* w1_out = (const float*)d_in[7];
  const float* b1_out = (const float*)d_in[8];
  const float* w2_in = (const float*)d_in[9];
  const float* b2_in = (const float*)d_in[10];
  const float* w2_hid = (const float*)d_in[11];
  const float* b2_hid = (const float*)d_in[12];
  const float* w2_out = (const float*)d_in[13];
  const float* b2_out = (const float*)d_in[14];
  const float* conv_in_w = (const float*)d_in[15];
  const float* conv_in_b = (const float*)d_in[16];
  const float* conv_hid_w = (const float*)d_in[17];
  const float* conv_hid_b = (const float*)d_in[18];
  const float* conv_out_w = (const float*)d_in[19];
  const float* conv_out_b = (const float*)d_in[20];
  const int* gs_part = (const int*)d_in[21];
  const int* uv_idx = (const int*)d_in[22];
  float* out = (float*)d_out;

  char* ws = (char*)d_ws;
  size_t off = 0;
  auto alloc = [&](size_t bytes) -> void* {
    void* p = ws + off;
    off = (off + bytes + 255) & ~(size_t)255;
    return p;
  };
  int* counts = (int*)alloc(NPART * 4);
  int* cursors = (int*)alloc(NPART * 4);
  int* offs = (int*)alloc((NPART + 1) * 4);
  int* tile_part = (int*)alloc(NT_TILES * 4);
  int* pidx = (int*)alloc((size_t)PIDX_LEN * 4);
  unsigned short* mlpw = (unsigned short*)alloc((size_t)NPART * MLPW_PART * 2);
  unsigned short* convw =
      (unsigned short*)alloc((size_t)(CONVIN_SW + 3 * CONVHID_SW) * 2);
  unsigned short* uvmap = (unsigned short*)alloc((size_t)65536 * 32 * 2);
  float* bufF = (float*)alloc((size_t)65536 * 64 * 4);
  unsigned short* bufB = (unsigned short*)alloc((size_t)65536 * 64 * 2);
  float* stats = (float*)alloc(128 * 4);

  init_kernel<<<8192, 256, 0, stream>>>(counts, cursors, tile_part, pidx, uvmap);
  mlp_prep_kernel<<<980, 512, 0, stream>>>(w1_in, w1_hid, w1_out, w2_in, w2_hid,
                                           w2_out, mlpw);
  conv_prep_kernel<<<252, 512, 0, stream>>>(conv_in_w, conv_hid_w, convw);
  count_kernel<<<157, 256, 0, stream>>>(gs_part, counts);
  offsets_kernel<<<1, 1, 0, stream>>>(counts, offs, tile_part);
  scatter_kernel<<<157, 256, 0, stream>>>(gs_part, offs, cursors, pidx);
  mlp_kernel<<<NT_TILES, 32, 0, stream>>>(tile_part, pidx, latent_z, latent_f,
                                          cano_xyz, mlpw, b1_in, b1_hid, b1_out,
                                          b2_in, b2_hid, b2_out, uv_idx, uvmap);
  // CNN: conv_in (32->64), then 3x { relu -> conv -> IN }, then conv_out+sigmoid
  conv_kernel<1><<<4096, 128, 0, stream>>>(uvmap, convw, conv_in_b, bufF);
  prep_kernel<<<16384, 256, 0, stream>>>(bufF, stats, /*relu only*/ 2, bufB);
  for (int l = 0; l < 3; ++l) {
    conv_kernel<2><<<4096, 128, 0, stream>>>(bufB, convw + CONVIN_SW + l * CONVHID_SW,
                                             conv_hid_b + l * 64, bufF);
    stats_kernel<<<64, 256, 0, stream>>>(bufF, stats);
    prep_kernel<<<16384, 256, 0, stream>>>(bufF, stats, (l < 2) ? 3 : 1, bufB);
  }
  convout_kernel<<<768, 256, 0, stream>>>(bufB, conv_out_w, conv_out_b, out);
}
